// TransitionModel_2980707303627
// MI455X (gfx1250) — matrix operand order repaired and checked by reading, not compiled, not run on hardware
//
#include <hip/hip_runtime.h>
#include <hip/hip_bf16.h>

// out[b,i] = LSE_k( log_softmax(W,axis=0)[i,k] + log_alpha[b,k] )
// Folded:  A'[b,k] = log_alpha[b,k] - colLSE(W)[k]
//          out[b,i] = m_w[i] + m_a[b] + log( sum_k P[i,k]*Q[b,k] )
//          P = exp(W - rowmax(W)) (f16), Q = exp(A' - rowmax(A')) (f16)
// GEMM S = Q x P^T done with v_wmma_f32_16x16x32_f16 (f32 accum).

typedef __attribute__((ext_vector_type(16))) _Float16 v16h;
typedef __attribute__((ext_vector_type(8)))  _Float16 v8h;
typedef __attribute__((ext_vector_type(8)))  float    v8f;

constexpr int N = 2048;   // state dim
constexpr int B = 128;    // batch
constexpr int KSEG = 4;   // K-split for the GEMM (occupancy), fixed-order combine
constexpr int KSEG_LEN = N / KSEG;  // 512

union FragA { v16h v; v8h h[2]; };

// ---------------------------------------------------------------- kernel 1
// colLSE[k] = logsumexp_i W[i,k].  One lane per column; loads are coalesced
// across lanes (consecutive k). Branchless online LSE.
__global__ void col_lse_kernel(const float* __restrict__ W,
                               float* __restrict__ colLSE) {
  int k = blockIdx.x * blockDim.x + threadIdx.x;
  if (k >= N) return;
  float m = -__builtin_inff();
  float s = 0.0f;
  for (int i = 0; i < N; ++i) {
    float x  = W[(size_t)i * N + k];
    float nm = fmaxf(m, x);
    s = s * __expf(m - nm) + __expf(x - nm);
    m = nm;
  }
  colLSE[k] = m + __logf(s);
}

// ---------------------------------------------------------------- kernel 2
// Per row i of W: m_w[i] = max_k W[i,k];  P[i,k] = (f16) exp(W[i,k]-m_w[i]).
__global__ void __launch_bounds__(256)
rowmax_expP_kernel(const float* __restrict__ W,
                   _Float16* __restrict__ P,
                   float* __restrict__ m_w) {
  __shared__ float red[256];
  const int row = blockIdx.x;
  const int tid = threadIdx.x;
  const float* wr = W + (size_t)row * N;
  float v[8];
  float m = -__builtin_inff();
#pragma unroll
  for (int j = 0; j < 8; ++j) {
    v[j] = wr[tid + j * 256];
    m = fmaxf(m, v[j]);
  }
  red[tid] = m;
  __syncthreads();
  for (int off = 128; off > 0; off >>= 1) {
    if (tid < off) red[tid] = fmaxf(red[tid], red[tid + off]);
    __syncthreads();
  }
  m = red[0];
  if (tid == 0) m_w[row] = m;
  _Float16* pr = P + (size_t)row * N;
#pragma unroll
  for (int j = 0; j < 8; ++j)
    pr[tid + j * 256] = (_Float16)__expf(v[j] - m);
}

// ---------------------------------------------------------------- kernel 3
// Per batch row b: A'[b,k] = log_alpha[b,k] - colLSE[k];
// m_a[b] = max_k A';  Q[b,k] = (f16) exp(A' - m_a[b]).
__global__ void __launch_bounds__(256)
rowmax_expQ_kernel(const float* __restrict__ la,
                   const float* __restrict__ colLSE,
                   _Float16* __restrict__ Q,
                   float* __restrict__ m_a) {
  __shared__ float red[256];
  const int b   = blockIdx.x;
  const int tid = threadIdx.x;
  float v[8];
  float m = -__builtin_inff();
#pragma unroll
  for (int j = 0; j < 8; ++j) {
    int k = tid + j * 256;
    v[j] = la[(size_t)b * N + k] - colLSE[k];
    m = fmaxf(m, v[j]);
  }
  red[tid] = m;
  __syncthreads();
  for (int off = 128; off > 0; off >>= 1) {
    if (tid < off) red[tid] = fmaxf(red[tid], red[tid + off]);
    __syncthreads();
  }
  m = red[0];
  if (tid == 0) m_a[b] = m;
#pragma unroll
  for (int j = 0; j < 8; ++j)
    Q[(size_t)b * N + tid + j * 256] = (_Float16)__expf(v[j] - m);
}

// ---------------------------------------------------------------- kernel 4
// S = Q x P^T via WMMA. One wave handles one 16x16 output tile for one of
// KSEG K-segments; partial f32 sums go to workspace (fixed-order combine in
// the epilogue -> deterministic). 4096 waves total.
//
// A-frag (Q, 16x32 f16): lane l<16 -> M=l, halves = K{0..7,8..15};
//                        lane l>=16 -> M=l-16, halves = K{16..23,24..31}.
// B-frag (P^T, 32x16 f16): lane l<16 -> N=l, K = k0..k0+15 contiguous;
//                          lane l>=16 -> N=l-16, K = k0+16..k0+31.
// D: VGPR r -> M = 8*hi + r, N = lane&15.
__global__ void __launch_bounds__(256)
lse_gemm_kernel(const _Float16* __restrict__ Q,
                const _Float16* __restrict__ P,
                float* __restrict__ partial) {
  const int wave = (blockIdx.x * blockDim.x + threadIdx.x) >> 5;
  const int lane = threadIdx.x & 31;
  const int ks   = wave & (KSEG - 1);
  const int tile = wave >> 2;            // 0..1023
  const int tc   = tile & (N / 16 - 1);  // output column tile (i)
  const int tr   = tile >> 7;            // output row tile (b), 0..7
  const int l15  = lane & 15;
  const int hi   = lane >> 4;

  const _Float16* qrow = Q + (size_t)(tr * 16 + l15) * N;  // A: row = b
  const _Float16* prow = P + (size_t)(tc * 16 + l15) * N;  // B: "row" = i

  v8f c = {};
  const int kbeg = ks * KSEG_LEN;
#pragma unroll 4
  for (int k = kbeg; k < kbeg + KSEG_LEN; k += 32) {
    FragA a;
    a.h[0] = *(const v8h*)(qrow + k + hi * 16);       // K {0..7} / {16..23}
    a.h[1] = *(const v8h*)(qrow + k + hi * 16 + 8);   // K {8..15} / {24..31}
    v16h bf = *(const v16h*)(prow + k + hi * 16);     // K 16 contiguous
    c = __builtin_amdgcn_wmma_f32_16x16x32_f16(
        /*neg_a=*/false, a.v, /*neg_b=*/false, bf,
        /*c_mod=*/(short)0, c, /*reuse_a=*/false, /*reuse_b=*/false);
  }

  float* pout = partial +
      ((size_t)ks * B + (size_t)(tr * 16 + hi * 8)) * N + tc * 16 + l15;
#pragma unroll
  for (int r = 0; r < 8; ++r)
    pout[(size_t)r * N] = c[r];
}

// ---------------------------------------------------------------- kernel 5
// out[b,i] = log( sum_ks partial ) + m_a[b] + m_w[i]
__global__ void __launch_bounds__(256)
epilogue_kernel(const float* __restrict__ partial,
                const float* __restrict__ m_a,
                const float* __restrict__ m_w,
                float* __restrict__ out) {
  int idx = blockIdx.x * blockDim.x + threadIdx.x;  // b*N + i
  if (idx >= B * N) return;
  int b = idx >> 11;        // / N
  int i = idx & (N - 1);
  float s = 0.0f;
#pragma unroll
  for (int ks = 0; ks < KSEG; ++ks)
    s += partial[(size_t)ks * B * N + idx];
  out[idx] = __logf(s) + m_a[b] + m_w[i];
}

// ---------------------------------------------------------------- launch
extern "C" void kernel_launch(void* const* d_in, const int* in_sizes, int n_in,
                              void* d_out, int out_size, void* d_ws, size_t ws_size,
                              hipStream_t stream) {
  const float* log_alpha = (const float*)d_in[0];  // [B, N]
  const float* W         = (const float*)d_in[1];  // [N, N]
  float* out             = (float*)d_out;          // [B, N]

  // Workspace layout (bytes):
  //   0        colLSE   2048 f32 (8 KB)
  //   8192     m_w      2048 f32 (8 KB)
  //   16384    m_a       128 f32
  //   65536    Q        128x2048 f16 (512 KB)
  //   1 MB     P        2048x2048 f16 (8 MB)
  //   9 MB     partial  KSEG x 128 x 2048 f32 (4 MB)
  char* ws = (char*)d_ws;
  float*    colLSE  = (float*)(ws);
  float*    m_w     = (float*)(ws + 8192);
  float*    m_a     = (float*)(ws + 16384);
  _Float16* Q       = (_Float16*)(ws + 65536);
  _Float16* P       = (_Float16*)(ws + (size_t)(1u << 20));
  float*    partial = (float*)(ws + (size_t)9 * (1u << 20));

  col_lse_kernel<<<N / 256, 256, 0, stream>>>(W, colLSE);
  rowmax_expP_kernel<<<N, 256, 0, stream>>>(W, P, m_w);
  rowmax_expQ_kernel<<<B, 256, 0, stream>>>(log_alpha, colLSE, Q, m_a);

  const int gemm_waves  = KSEG * (B / 16) * (N / 16);   // 4096
  const int gemm_blocks = gemm_waves * 32 / 256;        // 512
  lse_gemm_kernel<<<gemm_blocks, 256, 0, stream>>>(Q, P, partial);

  epilogue_kernel<<<(B * N) / 256, 256, 0, stream>>>(partial, m_a, m_w, out);
}
